// TransMILPositionalEncoding_8014408974418
// MI455X (gfx1250) — matrix operand order repaired and checked by reading, not compiled, not run on hardware
//
#include <hip/hip_runtime.h>
#include <math.h>

typedef __attribute__((ext_vector_type(16))) __bf16 v16bf;
typedef __attribute__((ext_vector_type(16))) short  v16s;
typedef __attribute__((ext_vector_type(8)))  float  v8f;

// ---------------- model constants ----------------
#define BATCH  4
#define N0TOK  10000
#define SEQ    10001          // 1 cls + 10000
#define NP     10240          // padded nystrom length
#define PADF   239            // front pad
#define DMODEL 512
#define NH     8
#define DH     64
#define BHN    32             // BATCH*NH
#define MLM    256            // landmarks
#define LGRP   40             // tokens per landmark
#define CKW    33

#define LDSP   72             // LDS row stride in ushorts (144B = 9*16B)

// GEMM epilogue flags
#define EPI_RELU 1
#define EPI_AXPB 2            // C = scale*(alpha*E - A@B)

// ---------------- helpers ----------------
__device__ __forceinline__ unsigned short f2bf(float f) {
  unsigned int u = __float_as_uint(f);
  u += 0x7FFFu + ((u >> 16) & 1u);          // round-to-nearest-even
  return (unsigned short)(u >> 16);
}

// pack two fp32 -> packed bf16x2 (hardware packed convert when available)
__device__ __forceinline__ unsigned int f2bf2(float x, float y) {
#if defined(__has_builtin)
#if __has_builtin(__builtin_amdgcn_cvt_pk_bf16_f32)
  auto p = __builtin_amdgcn_cvt_pk_bf16_f32(x, y);
  return __builtin_bit_cast(unsigned int, p);
#else
  return (unsigned)f2bf(x) | ((unsigned)f2bf(y) << 16);
#endif
#else
  return (unsigned)f2bf(x) | ((unsigned)f2bf(y) << 16);
#endif
}

__device__ __forceinline__ float blk_sum(float v, float* red) {
  int t = threadIdx.x;
  red[t] = v; __syncthreads();
  for (int s = 128; s > 0; s >>= 1) {
    if (t < s) red[t] += red[t + s];
    __syncthreads();
  }
  float r = red[0]; __syncthreads();
  return r;
}

__device__ __forceinline__ float blk_max(float v, float* red) {
  int t = threadIdx.x;
  red[t] = v; __syncthreads();
  for (int s = 128; s > 0; s >>= 1) {
    if (t < s) red[t] = fmaxf(red[t], red[t + s]);
    __syncthreads();
  }
  float r = red[0]; __syncthreads();
  return r;
}

// ---------------- WMMA bf16 GEMM ----------------
// C[M,N] = epi( scale * A[M,K] @ op(B) )  batched over blockIdx.z
// btr==1: B is [N,K] row-major (weight layout, B^T applied)
// btr==0: B is [K,N] row-major
// flags: EPI_RELU -> relu;  EPI_AXPB -> C = scale*(alpha*E - A@B)
// bias != null: add bias[col]
// 128x128 block tile, K-tile 64, double-buffered LDS, 8 wave32 waves,
// each wave = 4x2 grid of 16x16 tiles -> 16 v_wmma per K-tile.
__global__ __launch_bounds__(256)
void gemm_wmma_bf16(const float* __restrict__ A, long long sA, int lda,
                    const float* __restrict__ B, long long sB, int ldb, int btr,
                    float* __restrict__ C, long long sC, int ldc,
                    const float* __restrict__ bias,
                    const float* __restrict__ E,      // extra operand for AXPB
                    int flags, float scale, float alpha,
                    int M, int N, int K)
{
  __shared__ unsigned short As[2][128][LDSP];
  __shared__ unsigned short Bs[2][128][LDSP];

  const int tid  = threadIdx.x;
  const int lane = tid & 31;
  const int wid  = tid >> 5;
  const int wm   = wid >> 2;     // 0..1  (64 rows each)
  const int wn   = wid & 3;      // 0..3  (32 cols each)

  const float* Ab = A + (long long)blockIdx.z * sA;
  const float* Bb = B + (long long)blockIdx.z * sB;
  float*       Cb = C + (long long)blockIdx.z * sC;
  const float* Eb = E ? (E + (long long)blockIdx.z * sC) : nullptr;

  const int m0 = blockIdx.y * 128;
  const int n0 = blockIdx.x * 128;

  float4 ra[8], rb[8];   // in-flight tile (held in VGPRs during compute)

  // issue global loads for the tile at k0 (128x64 of A, 64x128/128x64 of B)
  auto load_tile = [&](int k0) {
    const bool fa = (m0 + 128 <= M) && (k0 + 64 <= K);
    #pragma unroll
    for (int it = 0; it < 8; ++it) {
      int li = tid + it * 256;
      int rr = li >> 4, kc = (li & 15) * 4;
      if (fa) {
        ra[it] = *(const float4*)(Ab + (long long)(m0 + rr) * lda + k0 + kc);
      } else {
        float t[4];
        #pragma unroll
        for (int j = 0; j < 4; ++j) {
          int gm = m0 + rr, gk = k0 + kc + j;
          int cm = gm < M ? gm : M - 1;
          int cq = gk < K ? gk : K - 1;
          float v = Ab[(long long)cm * lda + cq];        // clamped, no exec dance
          t[j] = (gm < M && gk < K) ? v : 0.f;
        }
        ra[it] = make_float4(t[0], t[1], t[2], t[3]);
      }
    }
    if (btr) {
      const bool fb = (n0 + 128 <= N) && (k0 + 64 <= K);
      #pragma unroll
      for (int it = 0; it < 8; ++it) {
        int li = tid + it * 256;
        int nr = li >> 4, kc = (li & 15) * 4;
        if (fb) {
          rb[it] = *(const float4*)(Bb + (long long)(n0 + nr) * ldb + k0 + kc);
        } else {
          float t[4];
          #pragma unroll
          for (int j = 0; j < 4; ++j) {
            int gn = n0 + nr, gk = k0 + kc + j;
            int cn = gn < N ? gn : N - 1;
            int cq = gk < K ? gk : K - 1;
            float v = Bb[(long long)cn * ldb + cq];
            t[j] = (gn < N && gk < K) ? v : 0.f;
          }
          rb[it] = make_float4(t[0], t[1], t[2], t[3]);
        }
      }
    } else {
      const bool fb = (n0 + 128 <= N) && (k0 + 64 <= K);
      #pragma unroll
      for (int it = 0; it < 8; ++it) {
        int li = tid + it * 256;
        int kk = li >> 5, nc = (li & 31) * 4;
        if (fb) {
          rb[it] = *(const float4*)(Bb + (long long)(k0 + kk) * ldb + n0 + nc);
        } else {
          float t[4];
          #pragma unroll
          for (int j = 0; j < 4; ++j) {
            int gn = n0 + nc + j, gk = k0 + kk;
            int cn = gn < N ? gn : N - 1;
            int cq = gk < K ? gk : K - 1;
            float v = Bb[(long long)cq * ldb + cn];
            t[j] = (gn < N && gk < K) ? v : 0.f;
          }
          rb[it] = make_float4(t[0], t[1], t[2], t[3]);
        }
      }
    }
  };

  // convert fp32->bf16 and store the held tile into LDS buffer `buf`
  auto store_tile = [&](int buf) {
    #pragma unroll
    for (int it = 0; it < 8; ++it) {
      int li = tid + it * 256;
      int rr = li >> 4, kc = (li & 15) * 4;
      *(uint2*)&As[buf][rr][kc] =
          make_uint2(f2bf2(ra[it].x, ra[it].y), f2bf2(ra[it].z, ra[it].w));
    }
    if (btr) {
      #pragma unroll
      for (int it = 0; it < 8; ++it) {
        int li = tid + it * 256;
        int nr = li >> 4, kc = (li & 15) * 4;
        *(uint2*)&Bs[buf][nr][kc] =
            make_uint2(f2bf2(rb[it].x, rb[it].y), f2bf2(rb[it].z, rb[it].w));
      }
    } else {
      #pragma unroll
      for (int it = 0; it < 8; ++it) {
        int li = tid + it * 256;
        int kk = li >> 5, nc = (li & 31) * 4;
        Bs[buf][nc + 0][kk] = f2bf(rb[it].x);
        Bs[buf][nc + 1][kk] = f2bf(rb[it].y);
        Bs[buf][nc + 2][kk] = f2bf(rb[it].z);
        Bs[buf][nc + 3][kk] = f2bf(rb[it].w);
      }
    }
  };

  v8f acc[4][2];
  #pragma unroll
  for (int i = 0; i < 4; ++i)
    #pragma unroll
    for (int j = 0; j < 2; ++j)
      #pragma unroll
      for (int e = 0; e < 8; ++e) acc[i][j][e] = 0.f;

  const int nk = (K + 63) / 64;
  load_tile(0);
  store_tile(0);
  int buf = 0;

  for (int t = 0; t < nk; ++t) {
    __syncthreads();                       // staged tile `buf` is visible
    const bool more = (t + 1 < nk);
    if (more) load_tile((t + 1) * 64);     // global loads fly during WMMA below

    const int g   = lane >> 4;             // lane half
    const int r16 = lane & 15;

    #pragma unroll
    for (int ks = 0; ks < 64; ks += 32) {
      // B fragments: VGPR j holds K = ks + g*16 + 2j .. +1, N = lane&15
      v16bf bfrag[2];
      #pragma unroll
      for (int ni = 0; ni < 2; ++ni) {
        v16s tb;
        int nn = wn * 32 + ni * 16 + r16;
        #pragma unroll
        for (int vv = 0; vv < 8; ++vv) {
          int kk = ks + g * 16 + 2 * vv;
          tb[2 * vv]     = (short)Bs[buf][nn][kk];
          tb[2 * vv + 1] = (short)Bs[buf][nn][kk + 1];
        }
        bfrag[ni] = __builtin_bit_cast(v16bf, tb);
      }
      // A fragments: VGPR j holds K = ks + g*16 + 2j .. +1, M = lane&15
      #pragma unroll
      for (int mi = 0; mi < 4; ++mi) {
        v16s ta;
        int mm = wm * 64 + mi * 16 + r16;
        #pragma unroll
        for (int vv = 0; vv < 8; ++vv) {
          int kb = ks + g * 16 + 2 * vv;
          ta[2 * vv]     = (short)As[buf][mm][kb];
          ta[2 * vv + 1] = (short)As[buf][mm][kb + 1];
        }
        v16bf afrag = __builtin_bit_cast(v16bf, ta);
        #pragma unroll
        for (int ni = 0; ni < 2; ++ni) {
          acc[mi][ni] = __builtin_amdgcn_wmma_f32_16x16x32_bf16(
              false, afrag, false, bfrag[ni], (short)0, acc[mi][ni], false, false);
        }
      }
    }

    if (more) store_tile(buf ^ 1);         // convert + park next tile
    buf ^= 1;
  }

  // ---- epilogue: C/D layout: VGPR vr -> M = vr + (lane>=16 ? 8:0), N = lane&15
  const int gh  = lane >> 4;
  const int r16 = lane & 15;
  #pragma unroll
  for (int mi = 0; mi < 4; ++mi)
    #pragma unroll
    for (int ni = 0; ni < 2; ++ni)
      #pragma unroll
      for (int vr = 0; vr < 8; ++vr) {
        int row = m0 + wm * 64 + mi * 16 + gh * 8 + vr;
        int col = n0 + wn * 32 + ni * 16 + r16;
        if (row < M && col < N) {
          float v = acc[mi][ni][vr];
          if (flags & EPI_AXPB)
            v = scale * (alpha * Eb[(long long)row * ldc + col] - v);
          else
            v *= scale;
          if (bias) v += bias[col];
          if (flags & EPI_RELU) v = v > 0.f ? v : 0.f;
          Cb[(long long)row * ldc + col] = v;
        }
      }
}

// ---------------- elementwise / reduction kernels ----------------

__global__ __launch_bounds__(256)
void fill_cls(const float* __restrict__ cls, float* __restrict__ h)
{
  int idx = blockIdx.x * 256 + threadIdx.x;     // 4*512
  if (idx >= BATCH * DMODEL) return;
  int b = idx >> 9, d = idx & 511;
  h[(long long)b * SEQ * DMODEL + d] = cls[d];
}

// layernorm each token row, write into front-padded buffer [B,NP,D]
__global__ __launch_bounds__(256)
void ln_pad(const float* __restrict__ x, const float* __restrict__ w,
            const float* __restrict__ bb, float* __restrict__ y)
{
  __shared__ float red[256];
  int rid = blockIdx.x;                 // b*NP + i
  int b = rid / NP, i = rid % NP;
  float* yr = y + (long long)rid * DMODEL;
  if (i < PADF) {
    for (int d = threadIdx.x; d < DMODEL; d += 256) yr[d] = 0.f;
    return;
  }
  const float* xr = x + ((long long)b * SEQ + (i - PADF)) * DMODEL;
  float s = 0.f;
  for (int d = threadIdx.x; d < DMODEL; d += 256) s += xr[d];
  float mu = blk_sum(s, red) * (1.f / DMODEL);
  float vs = 0.f;
  for (int d = threadIdx.x; d < DMODEL; d += 256) {
    float dv = xr[d] - mu; vs += dv * dv;
  }
  float var = blk_sum(vs, red) * (1.f / DMODEL);
  float rstd = rsqrtf(var + 1e-5f);
  for (int d = threadIdx.x; d < DMODEL; d += 256)
    yr[d] = (xr[d] - mu) * rstd * w[d] + bb[d];
}

// qkv [B,NP,1536] -> q,k,v [B,H,NP,DH]; q scaled by dh^-0.5
__global__ __launch_bounds__(256)
void repack_qkv(const float* __restrict__ qkv, float* __restrict__ q,
                float* __restrict__ k, float* __restrict__ v, float scale)
{
  long long idx = (long long)blockIdx.x * 256 + threadIdx.x;   // B*NP*512
  if (idx >= (long long)BATCH * NP * DMODEL) return;
  int e  = (int)(idx & 63);
  int hh = (int)((idx >> 6) & 7);
  long long bn = idx >> 9;
  int n = (int)(bn % NP), b = (int)(bn / NP);
  long long src = bn * (3 * DMODEL) + hh * DH + e;
  long long dst = (((long long)b * NH + hh) * NP + n) * DH + e;
  q[dst] = qkv[src] * scale;
  k[dst] = qkv[src + DMODEL];
  v[dst] = qkv[src + 2 * DMODEL];
}

__global__ __launch_bounds__(256)
void landmark_mean(const float* __restrict__ q, const float* __restrict__ k,
                   float* __restrict__ ql, float* __restrict__ kl)
{
  long long idx = (long long)blockIdx.x * 256 + threadIdx.x;  // BH*MLM*DH
  if (idx >= (long long)BHN * MLM * DH) return;
  int e  = (int)(idx & 63);
  int lm = (int)((idx >> 6) & 255);
  int bh = (int)(idx >> 14);
  long long base = ((long long)bh * NP + (long long)lm * LGRP) * DH + e;
  float sq = 0.f, sk = 0.f;
  for (int j = 0; j < LGRP; ++j) {
    sq += q[base + (long long)j * DH];
    sk += k[base + (long long)j * DH];
  }
  ql[idx] = sq * (1.f / LGRP);
  kl[idx] = sk * (1.f / LGRP);
}

__global__ __launch_bounds__(256)
void softmax_rows(float* __restrict__ x, int cols)
{
  __shared__ float red[256];
  float* p = x + (long long)blockIdx.x * cols;
  float m = -3.4e38f;
  for (int c = threadIdx.x; c < cols; c += 256) m = fmaxf(m, p[c]);
  m = blk_max(m, red);
  float s = 0.f;
  for (int c = threadIdx.x; c < cols; c += 256) s += __expf(p[c] - m);
  s = blk_sum(s, red);
  float inv = 1.f / s;
  for (int c = threadIdx.x; c < cols; c += 256) p[c] = __expf(p[c] - m) * inv;
}

__global__ void zero_scal(float* scal)
{
  if (threadIdx.x < 2) scal[threadIdx.x] = 0.f;
}

// grid (BH*256, 2): y==0 row-sums max (sum over j), y==1 col-sums max
__global__ __launch_bounds__(256)
void pinv_norm(const float* __restrict__ x, float* __restrict__ scal)
{
  __shared__ float red[256];
  int bh = blockIdx.x >> 8;
  int i  = blockIdx.x & 255;
  const float* base = x + (long long)bh * MLM * MLM;
  float v = (blockIdx.y == 0) ? fabsf(base[i * MLM + threadIdx.x])
                              : fabsf(base[threadIdx.x * MLM + i]);
  v = blk_sum(v, red);
  if (threadIdx.x == 0)
    atomicMax(reinterpret_cast<unsigned int*>(scal + blockIdx.y),
              __float_as_uint(v));
}

// z0 = x^T / (rowmax * colmax)
__global__ __launch_bounds__(256)
void init_z(const float* __restrict__ x, float* __restrict__ z,
            const float* __restrict__ scal)
{
  long long idx = (long long)blockIdx.x * 256 + threadIdx.x;   // BH*256*256
  if (idx >= (long long)BHN * MLM * MLM) return;
  int j  = (int)(idx & 255);
  int i  = (int)((idx >> 8) & 255);
  int bh = (int)(idx >> 16);
  float inv = 1.f / (scal[0] * scal[1]);
  z[(long long)bh * MLM * MLM + (long long)j * MLM + i] =
      x[(long long)bh * MLM * MLM + (long long)i * MLM + j] * inv;
}

// out_heads += depthwise conv33 of v along sequence
__global__ __launch_bounds__(256)
void conv_residual(const float* __restrict__ v, const float* __restrict__ ck,
                   float* __restrict__ oh)
{
  long long idx = (long long)blockIdx.x * 256 + threadIdx.x;  // BH*NP*DH
  if (idx >= (long long)BHN * NP * DH) return;
  int e  = (int)(idx & 63);
  int i  = (int)((idx >> 6) % NP);
  int bh = (int)(idx / ((long long)NP * DH));
  int hh = bh & 7;
  long long rowbase = ((long long)bh * NP) * DH + e;
  float s = 0.f;
  #pragma unroll
  for (int j = 0; j < CKW; ++j) {
    int ii = i + j - CKW / 2;
    if (ii >= 0 && ii < NP) s += v[rowbase + (long long)ii * DH] * ck[hh * CKW + j];
  }
  oh[idx] += s;
}

// [B,H,NP,DH] -> [B,NP,512]
__global__ __launch_bounds__(256)
void merge_heads(const float* __restrict__ oh, float* __restrict__ attn)
{
  long long idx = (long long)blockIdx.x * 256 + threadIdx.x;  // B*NP*512
  if (idx >= (long long)BATCH * NP * DMODEL) return;
  int d = (int)(idx & 511);
  long long bn = idx >> 9;
  int n = (int)(bn % NP), b = (int)(bn / NP);
  int hh = d >> 6, e = d & 63;
  attn[idx] = oh[(((long long)b * NH + hh) * NP + n) * DH + e];
}

// h[b,t,:] += proj[b, t+PADF, :]  (+ sinusoidal pe(b,d) if add_pe)
__global__ __launch_bounds__(256)
void residual_add_pe(float* __restrict__ h, const float* __restrict__ proj,
                     int add_pe)
{
  long long idx = (long long)blockIdx.x * 256 + threadIdx.x;  // B*SEQ*512
  if (idx >= (long long)BATCH * SEQ * DMODEL) return;
  int d = (int)(idx % DMODEL);
  long long br = idx / DMODEL;
  int t = (int)(br % SEQ), b = (int)(br / SEQ);
  float add = proj[((long long)b * NP + t + PADF) * DMODEL + d];
  if (add_pe) {
    float div = __expf(-(float)(d & ~1) * (logf(10000.f) / (float)DMODEL));
    float ang = (float)b * div;
    add += (d & 1) ? __cosf(ang) : __sinf(ang);
  }
  h[idx] += add;
}

// final: LN(cls row) -> fc2 -> logits / softmax / argmax into d_out[20]
__global__ __launch_bounds__(256)
void final_head(const float* __restrict__ h, const float* __restrict__ w,
                const float* __restrict__ bb, const float* __restrict__ fw,
                const float* __restrict__ fb, float* __restrict__ out)
{
  __shared__ float red[256];
  int b = blockIdx.x;
  const float* xr = h + (long long)b * SEQ * DMODEL;
  float s = 0.f;
  for (int d = threadIdx.x; d < DMODEL; d += 256) s += xr[d];
  float mu = blk_sum(s, red) * (1.f / DMODEL);
  float vs = 0.f;
  for (int d = threadIdx.x; d < DMODEL; d += 256) {
    float dv = xr[d] - mu; vs += dv * dv;
  }
  float var = blk_sum(vs, red) * (1.f / DMODEL);
  float rstd = rsqrtf(var + 1e-5f);
  float d0 = 0.f, d1 = 0.f;
  for (int d = threadIdx.x; d < DMODEL; d += 256) {
    float nv = (xr[d] - mu) * rstd * w[d] + bb[d];
    d0 += nv * fw[d];
    d1 += nv * fw[DMODEL + d];
  }
  float l0 = blk_sum(d0, red);
  float l1 = blk_sum(d1, red);
  if (threadIdx.x == 0) {
    l0 += fb[0]; l1 += fb[1];
    float mx = fmaxf(l0, l1);
    float e0 = __expf(l0 - mx), e1 = __expf(l1 - mx);
    float inv = 1.f / (e0 + e1);
    out[b * 2 + 0] = l0;            out[b * 2 + 1] = l1;
    out[8 + b * 2 + 0] = e0 * inv;  out[8 + b * 2 + 1] = e1 * inv;
    out[16 + b] = (l1 > l0) ? 1.f : 0.f;
  }
}

// ---------------- host side ----------------
static inline void gemm(hipStream_t st,
                        const float* A, long long sA, int lda,
                        const float* B, long long sB, int ldb, int btr,
                        float* C, long long sC, int ldc,
                        const float* bias, const float* E,
                        int flags, float scale, float alpha,
                        int M, int N, int K, int batch)
{
  dim3 grid((N + 127) / 128, (M + 127) / 128, batch);
  gemm_wmma_bf16<<<grid, 256, 0, st>>>(A, sA, lda, B, sB, ldb, btr,
                                       C, sC, ldc, bias, E, flags, scale, alpha,
                                       M, N, K);
}

struct Ws {
  float *xn, *qkv, *q, *k, *v, *ql, *kl, *s1, *s3, *s2;
  float *z, *za, *xz, *t1, *t2, *a3v, *scal;
};

static void attention_layer(hipStream_t st, float* hbuf,
                            const float* nw, const float* nb,
                            const float* qkvw, const float* ow,
                            const float* ob, const float* ck,
                            const Ws& W, int add_pe)
{
  const long long EW = (long long)BHN * MLM * MLM;     // 32*256*256
  const long long SM = EW / BHN;                       // 256*256 per (b,h)

  // x_norm (front padded)
  ln_pad<<<BATCH * NP, 256, 0, st>>>(hbuf, nw, nb, W.xn);

  // qkv = xn @ qkv_w^T
  gemm(st, W.xn, 0, DMODEL, qkvw, 0, DMODEL, 1, W.qkv, 0, 3 * DMODEL,
       nullptr, nullptr, 0, 1.f, 0.f, BATCH * NP, 3 * DMODEL, DMODEL, 1);

  long long tot = (long long)BATCH * NP * DMODEL;
  repack_qkv<<<(int)((tot + 255) / 256), 256, 0, st>>>(W.qkv, W.q, W.k, W.v, 0.125f);

  long long lmt = (long long)BHN * MLM * DH;
  landmark_mean<<<(int)((lmt + 255) / 256), 256, 0, st>>>(W.q, W.k, W.ql, W.kl);

  // sim1 = q @ kl^T   [NP,256]
  gemm(st, W.q, (long long)NP * DH, DH, W.kl, (long long)MLM * DH, DH, 1,
       W.s1, (long long)NP * MLM, MLM, nullptr, nullptr, 0, 1.f, 0.f,
       NP, MLM, DH, BHN);
  // sim2 = ql @ kl^T  [256,256]
  gemm(st, W.ql, (long long)MLM * DH, DH, W.kl, (long long)MLM * DH, DH, 1,
       W.s2, SM, MLM, nullptr, nullptr, 0, 1.f, 0.f, MLM, MLM, DH, BHN);
  // sim3 = ql @ k^T   [256,NP]
  gemm(st, W.ql, (long long)MLM * DH, DH, W.k, (long long)NP * DH, DH, 1,
       W.s3, (long long)MLM * NP, NP, nullptr, nullptr, 0, 1.f, 0.f,
       MLM, NP, DH, BHN);

  softmax_rows<<<BHN * NP, 256, 0, st>>>(W.s1, MLM);
  softmax_rows<<<BHN * MLM, 256, 0, st>>>(W.s2, MLM);
  softmax_rows<<<BHN * MLM, 256, 0, st>>>(W.s3, NP);

  // Moore-Penrose pinv of a2 (s2), 6 iterations; the (alpha*I - X) factors are
  // folded into GEMM epilogues:  X@(aI - Y) == a*X - X@Y  (AXPB epilogue)
  zero_scal<<<1, 32, 0, st>>>(W.scal);
  pinv_norm<<<dim3(BHN * MLM, 2), 256, 0, st>>>(W.s2, W.scal);
  init_z<<<(int)((EW + 255) / 256), 256, 0, st>>>(W.s2, W.z, W.scal);

  float* z  = W.z;
  float* za = W.za;
  for (int it = 0; it < 6; ++it) {
    // xz = x @ z
    gemm(st, W.s2, SM, MLM, z, SM, MLM, 0, W.xz, SM, MLM,
         nullptr, nullptr, 0, 1.f, 0.f, MLM, MLM, MLM, BHN);
    // y1 = xz@(7I - xz) = 7*xz - xz@xz
    gemm(st, W.xz, SM, MLM, W.xz, SM, MLM, 0, W.t1, SM, MLM,
         nullptr, W.xz, EPI_AXPB, 1.f, 7.f, MLM, MLM, MLM, BHN);
    // y2 = xz@(15I - y1) = 15*xz - xz@y1
    gemm(st, W.xz, SM, MLM, W.t1, SM, MLM, 0, W.t2, SM, MLM,
         nullptr, W.xz, EPI_AXPB, 1.f, 15.f, MLM, MLM, MLM, BHN);
    // z' = 0.25*z@(13I - y2) = 0.25*(13*z - z@y2)
    gemm(st, z, SM, MLM, W.t2, SM, MLM, 0, za, SM, MLM,
         nullptr, z, EPI_AXPB, 0.25f, 13.f, MLM, MLM, MLM, BHN);
    float* tsw = z; z = za; za = tsw;
  }

  // a3v = a3 @ v   [256,DH]
  gemm(st, W.s3, (long long)MLM * NP, NP, W.v, (long long)NP * DH, DH, 0,
       W.a3v, (long long)MLM * DH, DH, nullptr, nullptr, 0, 1.f, 0.f,
       MLM, DH, NP, BHN);
  // tmp = a1 @ a2inv  [NP,256]  (overwrite s3, a3 is consumed)
  gemm(st, W.s1, (long long)NP * MLM, MLM, z, SM, MLM, 0,
       W.s3, (long long)NP * MLM, MLM, nullptr, nullptr, 0, 1.f, 0.f,
       NP, MLM, MLM, BHN);
  // out_heads = tmp @ a3v  [NP,DH]  (reuse q buffer)
  gemm(st, W.s3, (long long)NP * MLM, MLM, W.a3v, (long long)MLM * DH, DH, 0,
       W.q, (long long)NP * DH, DH, nullptr, nullptr, 0, 1.f, 0.f,
       NP, DH, MLM, BHN);

  long long cvt = (long long)BHN * NP * DH;
  conv_residual<<<(int)((cvt + 255) / 256), 256, 0, st>>>(W.v, ck, W.q);
  merge_heads<<<(int)((tot + 255) / 256), 256, 0, st>>>(W.q, W.qkv);

  // out projection (reuse xn as proj buffer)
  gemm(st, W.qkv, 0, DMODEL, ow, 0, DMODEL, 1, W.xn, 0, DMODEL,
       ob, nullptr, 0, 1.f, 0.f, BATCH * NP, DMODEL, DMODEL, 1);

  long long rt = (long long)BATCH * SEQ * DMODEL;
  residual_add_pe<<<(int)((rt + 255) / 256), 256, 0, st>>>(hbuf, W.xn, add_pe);
}

extern "C" void kernel_launch(void* const* d_in, const int* in_sizes, int n_in,
                              void* d_out, int out_size, void* d_ws, size_t ws_size,
                              hipStream_t stream)
{
  (void)in_sizes; (void)n_in; (void)out_size; (void)ws_size;
  const float* data    = (const float*)d_in[0];
  const float* fc1_w   = (const float*)d_in[1];
  const float* fc1_b   = (const float*)d_in[2];
  const float* cls_tok = (const float*)d_in[3];
  const float* l1_nw   = (const float*)d_in[4];
  const float* l1_nb   = (const float*)d_in[5];
  const float* l1_qkv  = (const float*)d_in[6];
  const float* l1_ow   = (const float*)d_in[7];
  const float* l1_ob   = (const float*)d_in[8];
  const float* l1_ck   = (const float*)d_in[9];
  const float* l2_nw   = (const float*)d_in[10];
  const float* l2_nb   = (const float*)d_in[11];
  const float* l2_qkv  = (const float*)d_in[12];
  const float* l2_ow   = (const float*)d_in[13];
  const float* l2_ob   = (const float*)d_in[14];
  const float* l2_ck   = (const float*)d_in[15];
  const float* norm_w  = (const float*)d_in[16];
  const float* norm_b  = (const float*)d_in[17];
  const float* fc2_w   = (const float*)d_in[18];
  const float* fc2_b   = (const float*)d_in[19];

  // bump allocator over d_ws
  float* base = (float*)d_ws;
  size_t off = 0;
  auto alloc = [&](long long n) {
    float* p = base + off;
    off += (size_t)((n + 63) & ~63LL);
    return p;
  };

  float* hbuf = alloc((long long)BATCH * SEQ * DMODEL);
  Ws W;
  W.xn   = alloc((long long)BATCH * NP * DMODEL);
  W.qkv  = alloc((long long)BATCH * NP * 3 * DMODEL);
  W.q    = alloc((long long)BATCH * NP * DMODEL);
  W.k    = alloc((long long)BATCH * NP * DMODEL);
  W.v    = alloc((long long)BATCH * NP * DMODEL);
  W.ql   = alloc((long long)BHN * MLM * DH);
  W.kl   = alloc((long long)BHN * MLM * DH);
  W.s1   = alloc((long long)BHN * NP * MLM);
  W.s3   = alloc((long long)BHN * MLM * NP);
  W.s2   = alloc((long long)BHN * MLM * MLM);
  W.z    = alloc((long long)BHN * MLM * MLM);
  W.za   = alloc((long long)BHN * MLM * MLM);
  W.xz   = alloc((long long)BHN * MLM * MLM);
  W.t1   = alloc((long long)BHN * MLM * MLM);
  W.t2   = alloc((long long)BHN * MLM * MLM);
  W.a3v  = alloc((long long)BHN * MLM * DH);
  W.scal = alloc(64);

  // h = relu(data @ fc1_w^T + b); cls token prepended
  fill_cls<<<(BATCH * DMODEL + 255) / 256, 256, 0, stream>>>(cls_tok, hbuf);
  gemm(stream, data, (long long)N0TOK * 1024, 1024,
       fc1_w, 0, 1024, 1,
       hbuf + DMODEL, (long long)SEQ * DMODEL, DMODEL,
       fc1_b, nullptr, EPI_RELU, 1.f, 0.f, N0TOK, DMODEL, 1024, BATCH);

  // two transformer layers (layer 1 folds the positional-encoding add)
  attention_layer(stream, hbuf, l1_nw, l1_nb, l1_qkv, l1_ow, l1_ob, l1_ck, W, 1);
  attention_layer(stream, hbuf, l2_nw, l2_nb, l2_qkv, l2_ow, l2_ob, l2_ck, W, 0);

  // classifier head -> logits[8], probs[8], yhat[4]
  final_head<<<BATCH, 256, 0, stream>>>(hbuf, norm_w, norm_b, fc2_w, fc2_b,
                                        (float*)d_out);
}
